// MoELayer_70025146794442
// MI455X (gfx1250) — matrix operand order repaired and checked by reading, not compiled, not run on hardware
//
#include <hip/hip_runtime.h>

// MoE layer for MI455X (gfx1250, wave32, WMMA bf16).
// Top-2 routed execution (2.7x fewer FLOPs than the dense reference math),
// bf16 weights (~85MB, L2-resident on 192MB L2), fp32 accumulation via
// v_wmma_f32_16x16x32_bf16. Down-projection A panels are double-buffered in
// LDS via global_load_async_to_lds_b128 (ASYNCcnt path).

#define DIMD 768
#define HIDD 2048
#define NEXP 8
#define NTOK 2048          // B*S = 4*512
#define TM   32            // token tile per block
#define TN   256           // N tile per block (8 waves: 2 M-rows x 4 N-cols x 64)
#define KP   256           // K panel staged in LDS
#define NTHREADS 256

typedef __attribute__((ext_vector_type(16))) __bf16 v16bf;
typedef __attribute__((ext_vector_type(8)))  float  v8f;

union BFrag {
    v16bf v;
    uint4 q[2];
    unsigned int u32[8];
    unsigned short u16[16];
};

__device__ __forceinline__ unsigned short f2bf(float f) {
    unsigned int u = __float_as_uint(f);
    u += 0x7FFFu + ((u >> 16) & 1u);      // round to nearest even
    return (unsigned short)(u >> 16);
}

__device__ __forceinline__ v8f wmma_bf16(const BFrag& a, const BFrag& b, v8f c) {
    return __builtin_amdgcn_wmma_f32_16x16x32_bf16(
        false, a.v, false, b.v, (short)0, c, false, false);
}

// A fragment (16x32 bf16): lane = M (lane&15), half = lane>>4.
// VGPR v: K = half*16 + 2v,(+1).
__device__ __forceinline__ void load_a_frag(const unsigned short* lds_row,
                                            int k0, int half, BFrag& a) {
#pragma unroll
    for (int v = 0; v < 8; ++v) {
        int kk = half * 16 + 2 * v;
        a.u32[v] = *(const unsigned int*)(lds_row + k0 + kk);
    }
}

// B fragment (32x16 bf16): lane -> N = lane&15, half = lane>>4,
// element j -> K = k0 + half*16 + j.  Weights stored TRANSPOSED [N][K] so this
// is one contiguous 32-byte per-lane load (two b128s).
__device__ __forceinline__ void load_b_fragT(const unsigned short* __restrict__ wT,
                                             int ktot, int k0, int half, int col,
                                             BFrag& b) {
    const uint4* p = (const uint4*)(wT + (size_t)col * ktot + k0 + half * 16);
    b.q[0] = p[0];
    b.q[1] = p[1];
}

// ---------------- weight conversion: fp32 [K][N] -> bf16 transposed [N][K] ---
__global__ __launch_bounds__(256) void k_cvt_t(const float* __restrict__ src,
                                               unsigned short* __restrict__ dst,
                                               int K, int N, int batch) {
    int i = blockIdx.x * 256 + threadIdx.x;
    int per = K * N;
    if (i >= per * batch) return;
    int b = i / per;
    int r = i - b * per;
    int k = r / N;
    int n = r - k * N;
    dst[(size_t)b * per + (size_t)n * K + k] = f2bf(src[i]);
}

// ---------------- router: softmax top-2, build per-expert token lists --------
__global__ __launch_bounds__(64) void k_router(const float* __restrict__ x,
                                               const int* __restrict__ loop_idx,
                                               const float* __restrict__ loop_emb,
                                               const float* __restrict__ router_w,
                                               int* __restrict__ counts,
                                               int* __restrict__ lists,
                                               float* __restrict__ wlists) {
    const int t = blockIdx.x;
    const int tid = threadIdx.x;
    const float* le = loop_emb + (size_t)loop_idx[0] * DIMD;

    float acc[NEXP];
#pragma unroll
    for (int e = 0; e < NEXP; ++e) acc[e] = 0.f;
    for (int d = tid; d < 2 * DIMD; d += 64) {
        const float v = (d < DIMD) ? x[(size_t)t * DIMD + d] : le[d - DIMD];
        const float* rw = router_w + (size_t)d * NEXP;
#pragma unroll
        for (int e = 0; e < NEXP; ++e) acc[e] += v * rw[e];
    }
    __shared__ float red[64 * NEXP];
    __shared__ float lg[NEXP];
#pragma unroll
    for (int e = 0; e < NEXP; ++e) red[tid * NEXP + e] = acc[e];
    __syncthreads();
    if (tid < NEXP) {
        float s = 0.f;
        for (int m = 0; m < 64; ++m) s += red[m * NEXP + tid];
        lg[tid] = s;
    }
    __syncthreads();
    if (tid == 0) {
        float mx = lg[0];
        for (int e = 1; e < NEXP; ++e) mx = fmaxf(mx, lg[e]);
        float p[NEXP], s = 0.f;
        for (int e = 0; e < NEXP; ++e) { p[e] = __expf(lg[e] - mx); s += p[e]; }
        int i0 = 0;
        for (int e = 1; e < NEXP; ++e) if (p[e] > p[i0]) i0 = e;
        int i1 = (i0 == 0) ? 1 : 0;
        for (int e = 0; e < NEXP; ++e) if (e != i0 && p[e] > p[i1]) i1 = e;
        const float v0 = p[i0] / s, v1 = p[i1] / s;
        const float inv = 1.f / (v0 + v1 + 1e-8f);
        int s0 = atomicAdd(&counts[i0], 1);
        lists[i0 * NTOK + s0] = t;  wlists[i0 * NTOK + s0] = v0 * inv;
        int s1 = atomicAdd(&counts[i1], 1);
        lists[i1 * NTOK + s1] = t;  wlists[i1 * NTOK + s1] = v1 * inv;
    }
}

__global__ void k_offsets(const int* __restrict__ counts, int* __restrict__ offsets) {
    if (threadIdx.x == 0) {
        int o = 0;
        for (int e = 0; e < NEXP; ++e) { offsets[e] = o; o += counts[e]; }
    }
}

// ---------------- gate/up GEMM + SwiGLU: h = silu(x@Wg) * (x@Wu) -------------
// grid.x: routed ? NEXP*(NTOK/TM) : NTOK/TM ; grid.y: HIDD/TN (=8)
__global__ __launch_bounds__(NTHREADS) void k_gateup(
        const float* __restrict__ x,
        const unsigned short* __restrict__ WgT0,   // bf16 transposed [HID][DIM]
        const unsigned short* __restrict__ WuT0,
        unsigned short* __restrict__ Hout,         // bf16 [rows][HID]
        const int* __restrict__ lists,
        const int* __restrict__ counts,
        const int* __restrict__ offsets,
        int routed) {
    const int tid = threadIdx.x;
    int nvalid, rowbase;
    const int* tok = nullptr;
    const unsigned short* WgT = WgT0;
    const unsigned short* WuT = WuT0;
    if (routed) {
        const int e = blockIdx.x >> 6;             // 64 tiles/expert
        const int tile = blockIdx.x & 63;
        const int cnt = counts[e];
        if (tile * TM >= cnt) return;              // uniform block exit
        nvalid = min(TM, cnt - tile * TM);
        tok = lists + e * NTOK + tile * TM;
        rowbase = offsets[e] + tile * TM;
        WgT = WgT0 + (size_t)e * DIMD * HIDD;
        WuT = WuT0 + (size_t)e * DIMD * HIDD;
    } else {
        nvalid = TM;
        rowbase = blockIdx.x * TM;
    }

    __shared__ unsigned short Alds[TM][KP + 8];

    const int lane = tid & 31, wv = tid >> 5;
    const int wrow = wv >> 2, wcol = wv & 3;
    const int half = lane >> 4, nl = lane & 15;
    const int m_off = wrow * 16;
    const int nbase = blockIdx.y * TN + wcol * 64;

    const v8f vzero = {0.f, 0.f, 0.f, 0.f, 0.f, 0.f, 0.f, 0.f};
    v8f accg[4], accu[4];
#pragma unroll
    for (int f = 0; f < 4; ++f) { accg[f] = vzero; accu[f] = vzero; }

    for (int p = 0; p < DIMD / KP; ++p) {
        // stage A panel (fp32 x -> bf16 LDS), zero-fill invalid rows
        for (int idx = tid; idx < TM * KP; idx += NTHREADS) {
            const int m = idx / KP;
            const int kk = idx - m * KP;
            float val = 0.f;
            if (m < nvalid) {
                const int t = tok ? tok[m] : (rowbase + m);
                val = x[(size_t)t * DIMD + p * KP + kk];
            }
            Alds[m][kk] = f2bf(val);
        }
        __syncthreads();
#pragma unroll
        for (int ks = 0; ks < KP / 32; ++ks) {
            BFrag a;
            load_a_frag(&Alds[m_off + nl][0], ks * 32, half, a);
            const int k0 = p * KP + ks * 32;
            if (k0 + 48 <= DIMD) {
                __builtin_prefetch(WgT + (size_t)(nbase + nl) * DIMD + k0 + 32, 0, 1);
                __builtin_prefetch(WuT + (size_t)(nbase + nl) * DIMD + k0 + 32, 0, 1);
            }
#pragma unroll
            for (int f = 0; f < 4; ++f) {
                BFrag bg, bu;
                load_b_fragT(WgT, DIMD, k0, half, nbase + f * 16 + nl, bg);
                load_b_fragT(WuT, DIMD, k0, half, nbase + f * 16 + nl, bu);
                accg[f] = wmma_bf16(a, bg, accg[f]);
                accu[f] = wmma_bf16(a, bu, accu[f]);
            }
        }
        __syncthreads();
    }

#pragma unroll
    for (int f = 0; f < 4; ++f) {
#pragma unroll
        for (int v = 0; v < 8; ++v) {
            const int m = m_off + v + 8 * half;         // C layout: M = v + 8*half
            if (m < nvalid) {
                const float g = accg[f][v];
                const float u = accu[f][v];
                const float h = (g / (1.f + __expf(-g))) * u;   // silu(g)*u
                Hout[(size_t)(rowbase + m) * HIDD + (nbase + f * 16 + nl)] = f2bf(h);
            }
        }
    }
}

// ---------------- down GEMM: out(+)= (w*) h @ Wd -----------------------------
// grid.x: routed ? NEXP*(NTOK/TM) : NTOK/TM ; grid.y: DIMD/TN (=3)
// A panels (bf16 h rows) double-buffered in LDS via async global->LDS DMA.
__global__ __launch_bounds__(NTHREADS) void k_down(
        const unsigned short* __restrict__ Hin,    // bf16 [rows][HID]
        const unsigned short* __restrict__ WdT0,   // bf16 transposed [DIM][HID]
        float* __restrict__ out,
        const int* __restrict__ lists,
        const float* __restrict__ wlists,
        const int* __restrict__ counts,
        const int* __restrict__ offsets,
        int routed) {
    const int tid = threadIdx.x;
    int nvalid, rowbase;
    const int* tok = nullptr;
    const float* wl = nullptr;
    const unsigned short* WdT = WdT0;
    if (routed) {
        const int e = blockIdx.x >> 6;
        const int tile = blockIdx.x & 63;
        const int cnt = counts[e];
        if (tile * TM >= cnt) return;
        nvalid = min(TM, cnt - tile * TM);
        tok = lists + e * NTOK + tile * TM;
        wl = wlists + e * NTOK + tile * TM;
        rowbase = offsets[e] + tile * TM;
        WdT = WdT0 + (size_t)e * HIDD * DIMD;
    } else {
        nvalid = TM;
        rowbase = blockIdx.x * TM;
    }

    __shared__ unsigned short Alds[2][TM][KP + 8];   // double-buffered panels

    const int lane = tid & 31, wv = tid >> 5;
    const int wrow = wv >> 2, wcol = wv & 3;
    const int half = lane >> 4, nl = lane & 15;
    const int m_off = wrow * 16;
    const int nbase = blockIdx.y * TN + wcol * 64;

    // Async-stage one K panel: 32 rows x 512B; each thread moves 4x16B chunks.
    // Rows >= nvalid are skipped (their LDS garbage only feeds masked-out
    // output rows, so no zero-fill is required).
    auto stage_async = [&](int p) {
        const int chunk = tid & 31;                  // 32 x 16B chunks per row
        const int mrow = tid >> 5;                   // 8 rows per pass
        unsigned short(*buf)[KP + 8] = Alds[p & 1];
#pragma unroll
        for (int pass = 0; pass < TM / 8; ++pass) {
            const int m = mrow + pass * 8;
            if (m < nvalid) {
                const unsigned short* gsrc =
                    Hin + (size_t)(rowbase + m) * HIDD + p * KP + chunk * 8;
                const unsigned int ldsoff =
                    (unsigned int)(unsigned long long)(&buf[m][0]) + chunk * 16u;
                asm volatile("global_load_async_to_lds_b128 %0, %1, off"
                             :: "v"(ldsoff), "v"(gsrc)
                             : "memory");
            }
        }
    };

    const v8f vzero = {0.f, 0.f, 0.f, 0.f, 0.f, 0.f, 0.f, 0.f};
    v8f acc[4];
#pragma unroll
    for (int f = 0; f < 4; ++f) acc[f] = vzero;

    stage_async(0);
    asm volatile("s_wait_asynccnt 0x0" ::: "memory");
    __syncthreads();

    for (int p = 0; p < HIDD / KP; ++p) {
        if (p + 1 < HIDD / KP) stage_async(p + 1);   // prefetch next panel
        const unsigned short* arow = &Alds[p & 1][m_off + nl][0];
#pragma unroll
        for (int ks = 0; ks < KP / 32; ++ks) {
            BFrag a;
            load_a_frag(arow, ks * 32, half, a);
            const int k0 = p * KP + ks * 32;
            if (k0 + 48 <= HIDD)
                __builtin_prefetch(WdT + (size_t)(nbase + nl) * HIDD + k0 + 32, 0, 1);
#pragma unroll
            for (int f = 0; f < 4; ++f) {
                BFrag b;
                load_b_fragT(WdT, HIDD, k0, half, nbase + f * 16 + nl, b);
                acc[f] = wmma_bf16(a, b, acc[f]);
            }
        }
        asm volatile("s_wait_asynccnt 0x0" ::: "memory");
        __syncthreads();
    }

#pragma unroll
    for (int f = 0; f < 4; ++f) {
#pragma unroll
        for (int v = 0; v < 8; ++v) {
            const int m = m_off + v + 8 * half;
            if (m < nvalid) {
                const int col = nbase + f * 16 + nl;
                if (routed) {
                    atomicAdd(&out[(size_t)tok[m] * DIMD + col], wl[m] * acc[f][v]);
                } else {
                    out[(size_t)(rowbase + m) * DIMD + col] = acc[f][v];
                }
            }
        }
    }
}

extern "C" void kernel_launch(void* const* d_in, const int* in_sizes, int n_in,
                              void* d_out, int out_size, void* d_ws, size_t ws_size,
                              hipStream_t stream) {
    (void)in_sizes; (void)n_in; (void)out_size; (void)ws_size;
    const float* x        = (const float*)d_in[0];
    const int*   loop_idx = (const int*)d_in[1];
    const float* sw_gate  = (const float*)d_in[2];
    const float* sw_up    = (const float*)d_in[3];
    const float* sw_down  = (const float*)d_in[4];
    const float* ewg      = (const float*)d_in[5];
    const float* ewu      = (const float*)d_in[6];
    const float* ewd      = (const float*)d_in[7];
    const float* loop_emb = (const float*)d_in[8];
    const float* router_w = (const float*)d_in[9];
    float* out = (float*)d_out;

    char* p = (char*)d_ws;
    auto alloc = [&](size_t bytes) {
        char* r = p;
        p += (bytes + 255) & ~(size_t)255;
        return r;
    };
    const size_t WSZ = (size_t)DIMD * HIDD;            // 1,572,864 elems
    unsigned short* WgS = (unsigned short*)alloc(WSZ * 2);
    unsigned short* WuS = (unsigned short*)alloc(WSZ * 2);
    unsigned short* WdS = (unsigned short*)alloc(WSZ * 2);
    unsigned short* EWG = (unsigned short*)alloc((size_t)NEXP * WSZ * 2);
    unsigned short* EWU = (unsigned short*)alloc((size_t)NEXP * WSZ * 2);
    unsigned short* EWD = (unsigned short*)alloc((size_t)NEXP * WSZ * 2);
    unsigned short* Hsh = (unsigned short*)alloc((size_t)NTOK * HIDD * 2);
    unsigned short* Hrt = (unsigned short*)alloc((size_t)2 * NTOK * HIDD * 2);
    int*   counts  = (int*)alloc(NEXP * sizeof(int));
    int*   offsets = (int*)alloc(NEXP * sizeof(int));
    int*   lists   = (int*)alloc((size_t)NEXP * NTOK * sizeof(int));
    float* wlists  = (float*)alloc((size_t)NEXP * NTOK * sizeof(float));

    hipMemsetAsync(counts, 0, NEXP * sizeof(int), stream);

    // weight conversion fp32 [K][N] -> bf16 transposed [N][K]
    auto cvt = [&](const float* s, unsigned short* d, int K, int N, int batch) {
        const int n = K * N * batch;
        k_cvt_t<<<dim3((n + 255) / 256), dim3(256), 0, stream>>>(s, d, K, N, batch);
    };
    cvt(sw_gate, WgS, DIMD, HIDD, 1);
    cvt(sw_up,   WuS, DIMD, HIDD, 1);
    cvt(sw_down, WdS, HIDD, DIMD, 1);
    cvt(ewg, EWG, DIMD, HIDD, NEXP);
    cvt(ewu, EWU, DIMD, HIDD, NEXP);
    cvt(ewd, EWD, HIDD, DIMD, NEXP);

    k_router<<<dim3(NTOK), dim3(64), 0, stream>>>(x, loop_idx, loop_emb, router_w,
                                                  counts, lists, wlists);
    k_offsets<<<dim3(1), dim3(32), 0, stream>>>(counts, offsets);

    // shared expert
    k_gateup<<<dim3(NTOK / TM, HIDD / TN), dim3(NTHREADS), 0, stream>>>(
        x, WgS, WuS, Hsh, nullptr, nullptr, nullptr, 0);
    // routed experts (gathered tokens)
    k_gateup<<<dim3(NEXP * (NTOK / TM), HIDD / TN), dim3(NTHREADS), 0, stream>>>(
        x, EWG, EWU, Hrt, lists, counts, offsets, 1);

    // down projections: shared writes base, routed accumulates
    k_down<<<dim3(NTOK / TM, DIMD / TN), dim3(NTHREADS), 0, stream>>>(
        Hsh, WdS, out, nullptr, nullptr, nullptr, nullptr, 0);
    k_down<<<dim3(NEXP * (NTOK / TM), DIMD / TN), dim3(NTHREADS), 0, stream>>>(
        Hrt, EWD, out, lists, wlists, counts, offsets, 1);
}
